// CausalSelfAttention_48120813584707
// MI455X (gfx1250) — matrix operand order repaired and checked by reading, not compiled, not run on hardware
//
#include <hip/hip_runtime.h>
#include <hip/hip_bf16.h>

typedef __bf16 bf16;
typedef __bf16 bf16x2 __attribute__((ext_vector_type(2)));
typedef __bf16 bf16x4 __attribute__((ext_vector_type(4)));
typedef __bf16 bf16x8 __attribute__((ext_vector_type(8)));
typedef __bf16 v16bf  __attribute__((ext_vector_type(16)));
typedef float  v8f    __attribute__((ext_vector_type(8)));
typedef float  v4f    __attribute__((ext_vector_type(4)));

#define B_ 2
#define T_ 2048
#define C_ 1024
#define H_ 16
#define D_ 64

#define WMMA_BF16(a, b, c) \
  __builtin_amdgcn_wmma_f32_16x16x32_bf16(false, (a), false, (b), (short)0, (c), false, false)

static __device__ __forceinline__ v8f zero8() {
  v8f z = {0.f, 0.f, 0.f, 0.f, 0.f, 0.f, 0.f, 0.f};
  return z;
}

// A fragment (16x32 bf16, MxK). rowptr = per-lane pointer to (row M = lane&15, k=0).
// ISA 7.12.2: lanes 0-15: VGPR j<4 -> K=2j,2j+1 ; j>=4 -> K=2j+8,2j+9. lanes 16-31: +8.
static __device__ __forceinline__ v16bf load_frag_a(const bf16* rowptr) {
  const int hi = (threadIdx.x & 16) ? 8 : 0;
  v16bf a;
#pragma unroll
  for (int j = 0; j < 8; ++j) {
    const int kb = ((j < 4) ? (2 * j) : (2 * j + 8)) + hi;
    bf16x2 p = *(const bf16x2*)(rowptr + kb);
    a[2 * j + 0] = p[0];
    a[2 * j + 1] = p[1];
  }
  return a;
}

// B fragment (32x16 bf16, KxN). colptr = per-lane pointer to (col N = lane&15, k=0)
// in a K-contiguous layout. lanes 0-15: j<4 -> K=2j,2j+1 ; j>=4 -> K=2j+8,2j+9. lanes 16-31: +8.
static __device__ __forceinline__ v16bf load_frag_b(const bf16* colptr) {
  const int hi = (threadIdx.x & 16) ? 8 : 0;
  v16bf b;
#pragma unroll
  for (int j = 0; j < 8; ++j) {
    const int kb = ((j < 4) ? (2 * j) : (2 * j + 8)) + hi;
    bf16x2 p = *(const bf16x2*)(colptr + kb);
    b[2 * j + 0] = p[0];
    b[2 * j + 1] = p[1];
  }
  return b;
}

// ---------------------------------------------------------------------------
// fp32 -> bf16 elementwise conversion (vectorized x4)
// ---------------------------------------------------------------------------
__global__ void f32_to_bf16_kernel(const float* __restrict__ src,
                                   bf16* __restrict__ dst, int n4) {
  int i = blockIdx.x * blockDim.x + threadIdx.x;
  const int stride = gridDim.x * blockDim.x;
  for (; i < n4; i += stride) {
    v4f s = *(const v4f*)(src + (size_t)i * 4);
    bf16x4 d;
#pragma unroll
    for (int u = 0; u < 4; ++u) d[u] = (bf16)s[u];
    *(bf16x4*)(dst + (size_t)i * 4) = d;
  }
}

// ---------------------------------------------------------------------------
// bf16 GEMM: C[M,N] = A[M,K] * B[K,N], fp32 accumulate.
// Block tile 128x128, 8 waves (2x4), wave tile 64x32 (4x2 of 16x16).
// K-stage 64 (two WMMA k-steps per barrier), double-buffered LDS with
// register staging so global loads for stage s+1 overlap WMMAs of stage s.
// M,N multiples of 128; K multiple of 64.
// ---------------------------------------------------------------------------
template <bool BF16_OUT>
__global__ __launch_bounds__(256) void gemm_bf16_kernel(
    const bf16* __restrict__ A, const bf16* __restrict__ Bm,
    int M, int N, int K, void* __restrict__ Cout) {
  __shared__ bf16 Alds[2][128][72];   // [buf][m][k], padded
  __shared__ bf16 BldsT[2][128][72];  // [buf][n][k], K-pairs contiguous

  const int tid  = threadIdx.x;
  const int ln   = tid & 31;
  const int ln15 = ln & 15;
  const int wave = tid >> 5;
  const int wm   = wave >> 2;  // 0..1
  const int wn   = wave & 3;   // 0..3
  const int m0   = blockIdx.y * 128;
  const int n0   = blockIdx.x * 128;

  v8f acc[4][2];
#pragma unroll
  for (int mt = 0; mt < 4; ++mt)
#pragma unroll
    for (int nt = 0; nt < 2; ++nt) acc[mt][nt] = zero8();

  const int KT = K >> 6;  // number of 64-wide K stages

  // --- prologue: stage 0 straight into buffer 0 ---
  {
#pragma unroll
    for (int it = 0; it < 4; ++it) {
      const int idx = it * 2048 + tid * 8;
      const int ar = idx >> 6, ac = idx & 63;
      bf16x8 av = *(const bf16x8*)(A + (size_t)(m0 + ar) * K + ac);
      *(bf16x8*)&Alds[0][ar][ac] = av;
      const int br = idx >> 7, bc = idx & 127;
      bf16x8 bv = *(const bf16x8*)(Bm + (size_t)br * N + n0 + bc);
#pragma unroll
      for (int u = 0; u < 8; ++u) BldsT[0][bc + u][br] = bv[u];
    }
  }
  __syncthreads();

  for (int kt = 0; kt < KT; ++kt) {
    const int cur = kt & 1;
    const bool has_next = (kt + 1 < KT);
    const int k1 = (kt + 1) << 6;

    // issue global loads for next stage (latency hidden under WMMAs below)
    bf16x8 ra[4], rb[4];
    if (has_next) {
#pragma unroll
      for (int it = 0; it < 4; ++it) {
        const int idx = it * 2048 + tid * 8;
        const int ar = idx >> 6, ac = idx & 63;
        ra[it] = *(const bf16x8*)(A + (size_t)(m0 + ar) * K + k1 + ac);
        const int br = idx >> 7, bc = idx & 127;
        rb[it] = *(const bf16x8*)(Bm + (size_t)(k1 + br) * N + n0 + bc);
      }
    }

    // compute on current buffer: 2 k-steps x (4 m x 2 n) = 16 WMMAs
#pragma unroll
    for (int kk = 0; kk < 2; ++kk) {
      v16bf af[4];
#pragma unroll
      for (int mt = 0; mt < 4; ++mt)
        af[mt] = load_frag_a(&Alds[cur][wm * 64 + mt * 16 + ln15][kk * 32]);
      v16bf bfr[2];
#pragma unroll
      for (int nt = 0; nt < 2; ++nt)
        bfr[nt] = load_frag_b(&BldsT[cur][wn * 32 + nt * 16 + ln15][kk * 32]);
#pragma unroll
      for (int mt = 0; mt < 4; ++mt)
#pragma unroll
        for (int nt = 0; nt < 2; ++nt)
          acc[mt][nt] = WMMA_BF16(af[mt], bfr[nt], acc[mt][nt]);
    }

    // commit staged registers into the other buffer
    if (has_next) {
      const int nb = cur ^ 1;
#pragma unroll
      for (int it = 0; it < 4; ++it) {
        const int idx = it * 2048 + tid * 8;
        const int ar = idx >> 6, ac = idx & 63;
        *(bf16x8*)&Alds[nb][ar][ac] = ra[it];
        const int br = idx >> 7, bc = idx & 127;
#pragma unroll
        for (int u = 0; u < 8; ++u) BldsT[nb][bc + u][br] = rb[it][u];
      }
    }
    __syncthreads();
  }

  // --- branch-free epilogue ---
  const int hi8 = (ln & 16) ? 8 : 0;
#pragma unroll
  for (int mt = 0; mt < 4; ++mt)
#pragma unroll
    for (int nt = 0; nt < 2; ++nt) {
      const size_t base =
          (size_t)(m0 + wm * 64 + mt * 16 + hi8) * N + (n0 + wn * 32 + nt * 16 + ln15);
      if constexpr (BF16_OUT) {
        bf16* Cb = (bf16*)Cout;
#pragma unroll
        for (int j = 0; j < 8; ++j) Cb[base + (size_t)j * N] = (bf16)acc[mt][nt][j];
      } else {
        float* Cf = (float*)Cout;
#pragma unroll
        for (int j = 0; j < 8; ++j) Cf[base + (size_t)j * N] = acc[mt][nt][j];
      }
    }
}

// ---------------------------------------------------------------------------
// qkv [B,T,3C] bf16 -> Qh/Kh/Vh [B*H, T, D] bf16, Q scaled by 1/sqrt(D)=0.125.
// Vectorized: one bf16x8 chunk (8 consecutive d within one head) per step.
// ---------------------------------------------------------------------------
__global__ void qkv_rearrange_kernel(const bf16* __restrict__ qkv,
                                     bf16* __restrict__ Qh,
                                     bf16* __restrict__ Kh,
                                     bf16* __restrict__ Vh) {
  int i = blockIdx.x * blockDim.x + threadIdx.x;
  const int n8 = B_ * T_ * C_ / 8;
  const int stride = gridDim.x * blockDim.x;
  for (; i < n8; i += stride) {
    const int g   = i * 8;
    const int bt  = g >> 10;      // C_ = 1024
    const int col = g & 1023;     // multiple of 8
    const int h = col >> 6, d = col & 63;
    const int b = bt >> 11;       // T_ = 2048
    const int t = bt & 2047;
    const size_t src = (size_t)bt * (3 * C_) + col;
    const size_t dst = (((size_t)b * H_ + h) * T_ + t) * D_ + d;

    bf16x8 qv = *(const bf16x8*)(qkv + src);
#pragma unroll
    for (int u = 0; u < 8; ++u) qv[u] = (bf16)((float)qv[u] * 0.125f);
    *(bf16x8*)(Qh + dst) = qv;
    *(bf16x8*)(Kh + dst) = *(const bf16x8*)(qkv + src + C_);
    *(bf16x8*)(Vh + dst) = *(const bf16x8*)(qkv + src + 2 * C_);
  }
}

// ---------------------------------------------------------------------------
// Flash attention: one block per (b*h, 64-query tile). 4 waves, each owns 16
// query rows. Double-buffered K/V tiles with register staging; online softmax
// in fp32 with 2 lanes per row (shfl_xor combine); WMMA bf16 for S and P*V.
// ---------------------------------------------------------------------------
__global__ __launch_bounds__(128) void attn_kernel(
    const bf16* __restrict__ Qh, const bf16* __restrict__ Kh,
    const bf16* __restrict__ Vh, bf16* __restrict__ Yb) {
  __shared__ bf16  Klds[2][64][72];   // [buf][key][d]
  __shared__ bf16  VldsT[2][64][72];  // [buf][d][key]
  __shared__ float Slds[64][68];      // scores fp32
  __shared__ bf16  Plds[64][72];      // probabilities bf16 [row][key]
  __shared__ float rowTmp[64];        // alpha / 1/l broadcast

  const int tid  = threadIdx.x;
  const int ln   = tid & 31;
  const int ln15 = ln & 15;
  const int w    = tid >> 5;  // 0..3
  const int hi8  = (ln & 16) ? 8 : 0;

  // softmax mapping: lanes 2r,2r+1 cooperate on row r
  const int srow  = tid >> 1;         // 0..63
  const int shalf = (tid & 1) * 32;   // column base 0 or 32

  const int qt = blockIdx.x;  // 0..T/64-1
  const int bh = blockIdx.y;  // 0..B*H-1
  const int b  = bh >> 4;     // H_ = 16
  const int h  = bh & 15;
  const int qBase = qt * 64;

  // Q fragments (rows qBase + w*16 .. +15), 1/sqrt(D) already folded in.
  const bf16* qrow = Qh + ((size_t)bh * T_ + qBase + w * 16 + ln15) * D_;
  v16bf qf[2];
  qf[0] = load_frag_a(qrow);
  qf[1] = load_frag_a(qrow + 32);

  v8f O[4];
#pragma unroll
  for (int nt = 0; nt < 4; ++nt) O[nt] = zero8();
  float m_i = -__builtin_inff();
  float l_i = 0.0f;

  // --- prologue: stage tile kt=0 into buffer 0 ---
  {
#pragma unroll
    for (int it = 0; it < 4; ++it) {
      const int idx = it * 1024 + tid * 8;
      const int r = idx >> 6, d0 = idx & 63;
      const size_t g = ((size_t)bh * T_ + r) * D_ + d0;
      bf16x8 kv = *(const bf16x8*)(Kh + g);
      *(bf16x8*)&Klds[0][r][d0] = kv;
      bf16x8 vv = *(const bf16x8*)(Vh + g);
#pragma unroll
      for (int u = 0; u < 8; ++u) VldsT[0][d0 + u][r] = vv[u];
    }
  }
  __syncthreads();

  for (int kt = 0; kt <= qt; ++kt) {
    const int cur = kt & 1;
    const bool has_next = (kt < qt);

    // issue global loads for next K/V tile (hidden under S/softmax/PV below)
    bf16x8 rk[4], rv[4];
    if (has_next) {
#pragma unroll
      for (int it = 0; it < 4; ++it) {
        const int idx = it * 1024 + tid * 8;
        const int r = idx >> 6, d0 = idx & 63;
        const size_t g = ((size_t)bh * T_ + (kt + 1) * 64 + r) * D_ + d0;
        rk[it] = *(const bf16x8*)(Kh + g);
        rv[it] = *(const bf16x8*)(Vh + g);
      }
    }

    // S = Q K^T  (contraction over d; B frag element (d, key) = Klds[key][d])
    v8f S[4];
#pragma unroll
    for (int nt = 0; nt < 4; ++nt) S[nt] = zero8();
#pragma unroll
    for (int kk = 0; kk < 2; ++kk)
#pragma unroll
      for (int nt = 0; nt < 4; ++nt) {
        v16bf bk = load_frag_b(&Klds[cur][nt * 16 + ln15][kk * 32]);
        S[nt] = WMMA_BF16(qf[kk], bk, S[nt]);
      }
#pragma unroll
    for (int nt = 0; nt < 4; ++nt)
#pragma unroll
      for (int j = 0; j < 8; ++j)
        Slds[w * 16 + j + hi8][nt * 16 + ln15] = S[nt][j];
    __syncthreads();

    // Online softmax: 2 lanes per row, 32 cols each; combine via shfl_xor(1).
    {
      const int kBound = (qBase + srow) - kt * 64;  // valid keys: c <= kBound
      float mrow = -__builtin_inff();
#pragma unroll
      for (int c4 = 0; c4 < 32; c4 += 4) {
        const int c = shalf + c4;
        v4f sv = *(const v4f*)&Slds[srow][c];
#pragma unroll
        for (int u = 0; u < 4; ++u) {
          float s = (c + u <= kBound) ? sv[u] : -__builtin_inff();
          mrow = fmaxf(mrow, s);
        }
      }
      mrow = fmaxf(mrow, __shfl_xor(mrow, 1));
      const float mnew  = fmaxf(m_i, mrow);
      const float alpha = __expf(m_i - mnew);
      float lsum = 0.0f;
#pragma unroll
      for (int c4 = 0; c4 < 32; c4 += 4) {
        const int c = shalf + c4;
        v4f sv = *(const v4f*)&Slds[srow][c];
        bf16x4 pk;
#pragma unroll
        for (int u = 0; u < 4; ++u) {
          float s = (c + u <= kBound) ? sv[u] : -__builtin_inff();
          float pv = __expf(s - mnew);
          lsum += pv;
          pk[u] = (bf16)pv;
        }
        *(bf16x4*)&Plds[srow][c] = pk;
      }
      lsum += __shfl_xor(lsum, 1);
      l_i = l_i * alpha + lsum;
      m_i = mnew;
      if (shalf == 0) rowTmp[srow] = alpha;
    }
    __syncthreads();

    // Rescale O by alpha per row, then O += P * V.
    float av[8];
#pragma unroll
    for (int j = 0; j < 8; ++j) av[j] = rowTmp[w * 16 + j + hi8];
#pragma unroll
    for (int nt = 0; nt < 4; ++nt)
#pragma unroll
      for (int j = 0; j < 8; ++j) O[nt][j] *= av[j];

#pragma unroll
    for (int kk = 0; kk < 2; ++kk) {
      v16bf pa = load_frag_a(&Plds[w * 16 + ln15][kk * 32]);
#pragma unroll
      for (int nt = 0; nt < 4; ++nt) {
        v16bf bv = load_frag_b(&VldsT[cur][nt * 16 + ln15][kk * 32]);
        O[nt] = WMMA_BF16(pa, bv, O[nt]);
      }
    }

    // commit staged K/V registers into the other buffer
    if (has_next) {
      const int nb = cur ^ 1;
#pragma unroll
      for (int it = 0; it < 4; ++it) {
        const int idx = it * 1024 + tid * 8;
        const int r = idx >> 6, d0 = idx & 63;
        *(bf16x8*)&Klds[nb][r][d0] = rk[it];
#pragma unroll
        for (int u = 0; u < 8; ++u) VldsT[nb][d0 + u][r] = rv[it][u];
      }
    }
    __syncthreads();
  }

  if ((tid & 1) == 0) rowTmp[srow] = (l_i > 0.0f) ? (1.0f / l_i) : 0.0f;
  __syncthreads();
  float lv[8];
#pragma unroll
  for (int j = 0; j < 8; ++j) lv[j] = rowTmp[w * 16 + j + hi8];
#pragma unroll
  for (int nt = 0; nt < 4; ++nt)
#pragma unroll
    for (int j = 0; j < 8; ++j) {
      const int t = qBase + w * 16 + j + hi8;
      const int d = nt * 16 + ln15;
      Yb[((size_t)b * T_ + t) * C_ + h * D_ + d] = (bf16)(O[nt][j] * lv[j]);
    }
}

// ---------------------------------------------------------------------------
// Launch
// ---------------------------------------------------------------------------
extern "C" void kernel_launch(void* const* d_in, const int* in_sizes, int n_in,
                              void* d_out, int out_size, void* d_ws, size_t ws_size,
                              hipStream_t stream) {
  const float* x      = (const float*)d_in[0];
  const float* W_attn = (const float*)d_in[1];
  const float* W_proj = (const float*)d_in[2];
  float* out = (float*)d_out;

  const size_t BT = (size_t)B_ * T_;  // 4096
  char* p = (char*)d_ws;
  bf16* xb   = (bf16*)p; p += BT * C_ * 2;                    // 8 MB
  bf16* Wab  = (bf16*)p; p += (size_t)C_ * 3 * C_ * 2;        // 6 MB
  bf16* Wpb  = (bf16*)p; p += (size_t)C_ * C_ * 2;            // 2 MB
  bf16* qkvb = (bf16*)p; p += BT * 3 * C_ * 2;                // 24 MB
  bf16* Qh   = (bf16*)p; p += (size_t)B_ * H_ * T_ * D_ * 2;  // 8 MB
  bf16* Kh   = (bf16*)p; p += (size_t)B_ * H_ * T_ * D_ * 2;  // 8 MB
  bf16* Vh   = (bf16*)p; p += (size_t)B_ * H_ * T_ * D_ * 2;  // 8 MB
  bf16* yb   = (bf16*)p; p += BT * C_ * 2;                    // 8 MB

  // 1) fp32 -> bf16 (vectorized x4)
  f32_to_bf16_kernel<<<2048, 256, 0, stream>>>(x, xb, (int)(BT * C_ / 4));
  f32_to_bf16_kernel<<<2048, 256, 0, stream>>>(W_attn, Wab, C_ * 3 * C_ / 4);
  f32_to_bf16_kernel<<<1024, 256, 0, stream>>>(W_proj, Wpb, C_ * C_ / 4);

  // 2) QKV GEMM: [4096,1024] x [1024,3072] -> bf16
  gemm_bf16_kernel<true><<<dim3(3 * C_ / 128, BT / 128), 256, 0, stream>>>(
      xb, Wab, (int)BT, 3 * C_, C_, (void*)qkvb);

  // 3) split + head-major rearrange (+ Q scale), vectorized x8
  qkv_rearrange_kernel<<<1024, 256, 0, stream>>>(qkvb, Qh, Kh, Vh);

  // 4) causal flash attention
  attn_kernel<<<dim3(T_ / 64, B_ * H_), 128, 0, stream>>>(Qh, Kh, Vh, yb);

  // 5) output projection: [4096,1024] x [1024,1024] -> fp32 d_out
  gemm_bf16_kernel<false><<<dim3(C_ / 128, BT / 128), 256, 0, stream>>>(
      yb, Wpb, (int)BT, C_, C_, (void*)out);
}
